// GRU_84043920048278
// MI455X (gfx1250) — compile-verified, matrix operand order checked by reading, not run on hardware
//
#include <hip/hip_runtime.h>

// ---------------- types for WMMA ----------------
typedef __attribute__((ext_vector_type(16))) __bf16       v16bf;
typedef __attribute__((ext_vector_type(8)))  float        v8f;
typedef __attribute__((ext_vector_type(4)))  unsigned int u32x4;

union Frag { u32x4 q[2]; v16bf v; };

// ---------------- problem constants ----------------
constexpr int Bb = 32, Ll = 1024, Ii = 1024, Hh = 1024;
constexpr int G3H = 3 * Hh;          // 3072
constexpr int NT  = G3H / 16;        // 192 column tiles over 3H
constexpr int KT  = Ii / 32;         // 32 k-tiles (depth 32 each)
constexpr int SLAB = 256;            // K columns staged in LDS per pass
constexpr int KT_SLAB = SLAB / 32;   // 8
constexpr int NSLAB = Ii / SLAB;     // 4
constexpr int NWG = 128;             // 64 layer-0 WGs + 64 layer-1 WGs
constexpr int NTHREADS = 384;        // 12 waves of 32

constexpr size_t WPACK_ELEMS = (size_t)4 * NT * KT * 512;  // 4 matrices, bf16
constexpr size_t WPACK_BYTES = WPACK_ELEMS * 2;            // 25,165,824 B
constexpr size_t H0_BYTES    = (size_t)2 * Bb * Hh * 4;    // double-buffered h0

// fp32 -> bf16, round-to-nearest-even
__device__ __forceinline__ unsigned short f2bf(float f) {
    unsigned u = __float_as_uint(f);
    u += 0x7FFFu + ((u >> 16) & 1u);
    return (unsigned short)(u >> 16);
}

// ---------------- barrier counter init ----------------
__global__ void gru_init_sync(unsigned* cnt) { *cnt = 0u; }

// ---------------- weight pre-pack: fp32 -> bf16 in WMMA-B fragment order ----
// element index o = ((mat*192 + ntile)*32 + ktile)*512 + lane*16 + j
// value  = W[layer][ kt*32 + j + 8*((j>>3)+(lane>>4)) ][ ntile*16 + (lane&15) ]
__global__ void gru_pack_weights(const float* __restrict__ Wi,
                                 const float* __restrict__ Wh,
                                 unsigned short* __restrict__ wpack) {
    int o    = blockIdx.x * 256 + threadIdx.x;     // exactly WPACK_ELEMS threads
    int j    = o & 15;
    int lane = (o >> 4) & 31;
    int kt   = (o >> 9) & 31;
    int rest = o >> 14;                            // mat*192 + nt
    int nt   = rest % NT;
    int mat  = rest / NT;                          // 0..3
    int layer = mat >> 1, which = mat & 1;         // which: 0=W_i, 1=W_h
    int k = kt * 32 + j + 8 * ((j >> 3) + (lane >> 4));
    int n = nt * 16 + (lane & 15);
    const float* W = which ? Wh : Wi;
    float v = W[(size_t)layer * Ii * G3H + (size_t)k * G3H + n];
    wpack[o] = f2bf(v);
}

// ---------------- device-wide barrier (persistent kernel) ----------------
__device__ __forceinline__ void grid_barrier(unsigned* cnt, unsigned target) {
    __syncthreads();
    if (threadIdx.x == 0) {
        __hip_atomic_fetch_add(cnt, 1u, __ATOMIC_RELEASE, __HIP_MEMORY_SCOPE_AGENT);
        while (__hip_atomic_load(cnt, __ATOMIC_ACQUIRE, __HIP_MEMORY_SCOPE_AGENT) < target) {
            __builtin_amdgcn_s_sleep(2);
        }
    }
    __syncthreads();
}

// ---------------- persistent GRU kernel ----------------
// WGs 0..63   : layer 0, hidden tile ht = wg        (computes h0[t=s])
// WGs 64..127 : layer 1, hidden tile ht = wg - 64   (computes h1[t=s-1])
__global__ __launch_bounds__(NTHREADS, 1)
void gru_persistent(const float* __restrict__ x,
                    const float* __restrict__ b_i,
                    const float* __restrict__ b_h,
                    const unsigned short* __restrict__ wpack,
                    float* __restrict__ h0buf,      // [2][32][1024]
                    unsigned* __restrict__ barcnt,
                    float* __restrict__ out) {      // [B][L][H] ++ [NL][B][H]
    __shared__ unsigned short sA[2][Bb][SLAB];      // bf16 stage: 32 KB
    __shared__ float sG[12][16][16];                // per-wave C tiles: 12 KB
    __shared__ float sBias[2][3][16];               // [bi/bh][gate][col]

    const int tid  = threadIdx.x;
    const int wg   = blockIdx.x;
    const int layerRole = wg >> 6;                  // 0 or 1
    const int ht   = wg & 63;                       // hidden tile (16 cols)
    const int wave = tid >> 5;
    const int lane = tid & 31;

    // per-wave GEMM role: matrix (gi vs gh), gate, M tile
    const int mw = wave / 6;                        // 0: input path, 1: recurrent
    const int g  = (wave % 6) >> 1;                 // 0=r, 1=z, 2=n
    const int mt = wave & 1;                        // batch-row tile
    const int ntile = g * 64 + ht;                  // column tile within 3H
    const int mat   = layerRole * 2 + mw;           // packed-weight matrix id
    const size_t tileBase = (size_t)(mat * NT + ntile) * KT;

    if (tid < 96) {                                 // preload biases for this tile
        int which = tid / 48, gg = (tid / 16) % 3, c = tid % 16;
        const float* bp = which ? b_h : b_i;
        sBias[which][gg][c] = bp[layerRole * G3H + gg * Hh + ht * 16 + c];
    }
    __syncthreads();

    for (int s = 0; s <= Ll; ++s) {
        const bool active = (layerRole == 0) ? (s < Ll) : (s >= 1);
        const int  t = (layerRole == 0) ? s : (s - 1);
        if (active) {
            // A-operand sources (rows = batch, cols = K)
            const float* srcA0; const float* srcA1;
            size_t str0, str1; bool zero1;
            if (layerRole == 0) {
                srcA0 = x + (size_t)t * Ii;          str0 = (size_t)Ll * Ii;   // x_t
                srcA1 = h0buf + ((s + 1) & 1) * (Bb * Hh); str1 = Hh;          // h0[t-1]
                zero1 = (t == 0);
            } else {
                srcA0 = h0buf + ((s - 1) & 1) * (Bb * Hh); str0 = Hh;          // h0[t]
                srcA1 = out + (size_t)(t > 0 ? t - 1 : 0) * Hh;                // h1[t-1]
                str1 = (size_t)Ll * Hh;
                zero1 = (t == 0);
            }

            v8f acc = {};
            for (int slab = 0; slab < NSLAB; ++slab) {
                __syncthreads();
                // cooperative bf16 staging of both A matrices for this K slab
                for (int e = tid; e < 2 * Bb * SLAB; e += NTHREADS) {
                    int within = e & (SLAB - 1);
                    int row    = (e >> 8) & 31;
                    int mtx    = e >> 13;
                    int col    = slab * SLAB + within;
                    float v;
                    if (mtx == 0) v = srcA0[(size_t)row * str0 + col];
                    else          v = zero1 ? 0.0f : srcA1[(size_t)row * str1 + col];
                    sA[mtx][row][within] = f2bf(v);
                }
                __syncthreads();

                if (slab + 1 < NSLAB)  // pull next slab's B fragments toward L2/L0
                    __builtin_prefetch(wpack + (tileBase + (size_t)(slab + 1) * KT_SLAB) * 512
                                             + (size_t)lane * 16, 0, 1);

                const int m     = mt * 16 + (lane & 15);
                const int khalf = 8 * (lane >> 4);
                #pragma unroll
                for (int kt = 0; kt < KT_SLAB; ++kt) {
                    Frag a, b;
                    const u32x4* pa = (const u32x4*)(&sA[mw][m][kt * 32 + khalf]);
                    a.q[0] = pa[0];                 // K = base .. base+7
                    a.q[1] = pa[2];                 // K = base+16 .. base+23
                    const u32x4* pb = (const u32x4*)(wpack
                        + (tileBase + (size_t)(slab * KT_SLAB + kt)) * 512
                        + (size_t)lane * 16);
                    b.q[0] = pb[0];
                    b.q[1] = pb[1];
                    acc = __builtin_amdgcn_wmma_f32_16x16x32_bf16(
                              false, a.v, false, b.v, (short)0, acc, false, false);
                }
            }
            __syncthreads();
            // spill the 16x16 C tile to LDS (C/D layout: VGPR r -> M=r(+8), lane -> N)
            {
                const int baseM = (lane >> 4) ? 8 : 0;
                #pragma unroll
                for (int r = 0; r < 8; ++r)
                    sG[wave][baseM + r][lane & 15] = acc[r];
            }
            __syncthreads();

            // fused gate math + state update, fp32
            for (int e = tid; e < Bb * 16; e += NTHREADS) {
                int row = e >> 4, col = e & 15;
                int mtl = row >> 4, r16 = row & 15;
                float gi_r = sG[0 + mtl][r16][col] + sBias[0][0][col];
                float gi_z = sG[2 + mtl][r16][col] + sBias[0][1][col];
                float gi_n = sG[4 + mtl][r16][col] + sBias[0][2][col];
                float gh_r = sG[6 + mtl][r16][col] + sBias[1][0][col];
                float gh_z = sG[8 + mtl][r16][col] + sBias[1][1][col];
                float gh_n = sG[10 + mtl][r16][col] + sBias[1][2][col];
                float rr = 1.0f / (1.0f + __expf(-(gi_r + gh_r)));
                float zz = 1.0f / (1.0f + __expf(-(gi_z + gh_z)));
                float xn = gi_n + rr * gh_n;
                float e2 = __expf(2.0f * xn);
                float nn = (e2 - 1.0f) / (e2 + 1.0f);        // tanh
                int c = ht * 16 + col;
                float hp, hn;
                if (layerRole == 0) {
                    hp = (t == 0) ? 0.0f
                        : h0buf[((s + 1) & 1) * (Bb * Hh) + row * Hh + c];
                    hn = zz * hp + (1.0f - zz) * nn;
                    h0buf[(s & 1) * (Bb * Hh) + row * Hh + c] = hn;
                    if (t == Ll - 1)                           // h_final[0]
                        out[(size_t)Bb * Ll * Hh + (size_t)row * Hh + c] = hn;
                } else {
                    hp = (t == 0) ? 0.0f
                        : out[(size_t)row * Ll * Hh + (size_t)(t - 1) * Hh + c];
                    hn = zz * hp + (1.0f - zz) * nn;
                    out[(size_t)row * Ll * Hh + (size_t)t * Hh + c] = hn;
                    if (t == Ll - 1)                           // h_final[1]
                        out[(size_t)Bb * Ll * Hh + (size_t)Bb * Hh
                            + (size_t)row * Hh + c] = hn;
                }
            }
        }
        grid_barrier(barcnt, (unsigned)NWG * (unsigned)(s + 1));
    }
}

extern "C" void kernel_launch(void* const* d_in, const int* in_sizes, int n_in,
                              void* d_out, int out_size, void* d_ws, size_t ws_size,
                              hipStream_t stream) {
    (void)in_sizes; (void)n_in; (void)out_size; (void)ws_size;
    const float* x   = (const float*)d_in[0];
    const float* W_i = (const float*)d_in[1];
    const float* b_i = (const float*)d_in[2];
    const float* W_h = (const float*)d_in[3];
    const float* b_h = (const float*)d_in[4];
    float* out = (float*)d_out;

    unsigned short* wpack = (unsigned short*)d_ws;
    float*    h0buf  = (float*)((char*)d_ws + WPACK_BYTES);
    unsigned* barcnt = (unsigned*)((char*)d_ws + WPACK_BYTES + H0_BYTES);

    gru_init_sync<<<1, 1, 0, stream>>>(barcnt);
    gru_pack_weights<<<(int)(WPACK_ELEMS / 256), 256, 0, stream>>>(W_i, W_h, wpack);
    gru_persistent<<<NWG, NTHREADS, 0, stream>>>(x, b_i, b_h, wpack, h0buf,
                                                 barcnt, out);
}
